// CustomRNNLayer_68143951118842
// MI455X (gfx1250) — matrix operand order repaired and checked by reading, not compiled, not run on hardware
//
#include <hip/hip_runtime.h>
#include <math.h>

// Problem constants (from reference): N=64, T=2048, D_IN=256, D_H=256
#define RNN_N   64
#define RNN_T   2048
#define RNN_DI  256
#define RNN_DH  256

typedef __attribute__((ext_vector_type(16))) __bf16          v16bf;
typedef __attribute__((ext_vector_type(8)))  float           v8f;
typedef __attribute__((ext_vector_type(8)))  unsigned short  ushort8;
typedef __attribute__((ext_vector_type(16))) unsigned short  ushort16;

// float -> bf16 with round-to-nearest-even (bit-level, no libcalls)
__device__ __forceinline__ unsigned short f2bf(float f) {
    unsigned int u = __builtin_bit_cast(unsigned int, f);
    unsigned int r = 0x7FFFu + ((u >> 16) & 1u);
    return (unsigned short)((u + r) >> 16);
}

__device__ __forceinline__ v16bf pack_frag(ushort8 lo, ushort8 hi) {
    ushort16 t = __builtin_shufflevector(lo, hi, 0,1,2,3,4,5,6,7,8,9,10,11,12,13,14,15);
    return __builtin_bit_cast(v16bf, t);
}

__device__ __forceinline__ v8f wmma_bf16(v16bf a, v16bf b, v8f c) {
    // 8 args: (neg_a, A, neg_b, B, c_mod, C, reuse_a, reuse_b)
    return __builtin_amdgcn_wmma_f32_16x16x32_bf16(false, a, false, b, (short)0, c, false, false);
}

// Native CDNA5 TRANS-unit tanh. Trailing v_nop satisfies the documented
// TRANS-op hazard (1 independent op before the result may be consumed).
__device__ __forceinline__ float fast_tanh(float x) {
    float y;
    asm volatile("v_tanh_f32 %0, %1\n\tv_nop" : "=v"(y) : "v"(x));
    return y;
}

// ---------------------------------------------------------------------------
// Kernel A: x_proj = seq_x @ Wx + bx  -> written into d_out (in-place staging)
// Block = 256 threads (8 waves). Block tile: 32 rows x 64 cols.
// Two independent half-K WMMA chains per wave for pipelining.
// ---------------------------------------------------------------------------
__global__ __launch_bounds__(256)
void xproj_kernel(const float* __restrict__ X, const float* __restrict__ Wx,
                  const float* __restrict__ bx, float* __restrict__ out) {
    __shared__ __align__(16) unsigned short As[32 * RNN_DI];      // [row][k]
    __shared__ __align__(16) unsigned short WsT[64 * RNN_DI];     // [n_local][k]

    const int bcol    = blockIdx.x & 3;        // 4 col-blocks of 64
    const int brow    = blockIdx.x >> 2;       // row-blocks of 32
    const int rowBase = brow * 32;             // row in flattened (N*T)
    const int colBase = bcol * 64;
    const int tid     = threadIdx.x;

    for (int idx = tid; idx < 32 * RNN_DI; idx += 256) {
        int r = idx >> 8, k = idx & 255;
        As[idx] = f2bf(X[(size_t)(rowBase + r) * RNN_DI + k]);
    }
    for (int idx = tid; idx < 64 * RNN_DI; idx += 256) {
        int k = idx >> 6, n = idx & 63;
        WsT[n * RNN_DI + k] = f2bf(Wx[(size_t)k * RNN_DH + colBase + n]);
    }
    __syncthreads();

    const int wave = tid >> 5, lane = tid & 31;
    const int rt   = wave >> 2;                // row tile 0..1
    const int ct   = wave & 3;                 // col tile 0..3
    const int m    = lane & 15;                // A row / B,C column in tile
    const int ks   = lane >> 4;                // K half selector
    const int ncol = colBase + ct * 16 + m;    // global output column
    const int nl   = ct * 16 + m;              // local column in WsT

    v8f ca, cb;
    {
        float b = bx[ncol];
        #pragma unroll
        for (int i = 0; i < 8; ++i) { ca[i] = b; cb[i] = 0.0f; }
    }

    const int arow = (rt * 16 + m) * RNN_DI;
    #pragma unroll
    for (int kb = 0; kb < 128; kb += 32) {
        // chain A: K = kb .. kb+31
        ushort8 alo = *reinterpret_cast<const ushort8*>(&As[arow + kb + ks * 16]);
        ushort8 ahi = *reinterpret_cast<const ushort8*>(&As[arow + kb + ks * 16 + 8]);
        ushort8 blo = *reinterpret_cast<const ushort8*>(&WsT[nl * RNN_DI + kb + ks * 16]);
        ushort8 bhi = *reinterpret_cast<const ushort8*>(&WsT[nl * RNN_DI + kb + ks * 16 + 8]);
        ca = wmma_bf16(pack_frag(alo, ahi), pack_frag(blo, bhi), ca);
        // chain B: K = kb+128 .. kb+159
        int kb2 = kb + 128;
        ushort8 clo = *reinterpret_cast<const ushort8*>(&As[arow + kb2 + ks * 16]);
        ushort8 chi = *reinterpret_cast<const ushort8*>(&As[arow + kb2 + ks * 16 + 8]);
        ushort8 dlo = *reinterpret_cast<const ushort8*>(&WsT[nl * RNN_DI + kb2 + ks * 16]);
        ushort8 dhi = *reinterpret_cast<const ushort8*>(&WsT[nl * RNN_DI + kb2 + ks * 16 + 8]);
        cb = wmma_bf16(pack_frag(clo, chi), pack_frag(dlo, dhi), cb);
    }

    // C layout: N = lane&15 ; VGPR r -> M = r + (lane>=16 ? 8 : 0)
    const int Mb = rt * 16 + ks * 8;
    #pragma unroll
    for (int r = 0; r < 8; ++r) {
        out[(size_t)(rowBase + Mb + r) * RNN_DH + ncol] = ca[r] + cb[r];
    }
}

// ---------------------------------------------------------------------------
// Kernel B: sequential recurrence  h = tanh(xp_t + h @ Wh + bh)
// 4 blocks x 16 batch rows. WhT (bf16) + double-buffered h state in LDS.
// Each of 8 waves owns a 16x32 column slice; 4 independent WMMA chains.
// xp read in-place from d_out (overlapped with the WMMA chain), then
// overwritten with h. One barrier per timestep (double buffer).
// ---------------------------------------------------------------------------
__global__ __launch_bounds__(256)
void rnn_kernel(const float* __restrict__ Wh, const float* __restrict__ bh,
                const float* __restrict__ hinit, float* __restrict__ out) {
    __shared__ __align__(16) unsigned short WhT[RNN_DH * RNN_DH];  // 128 KB [n][k]
    __shared__ __align__(16) unsigned short hS[2][16 * RNN_DH];    //  16 KB [buf][m][k]

    const int tid     = threadIdx.x;
    const int rowBase = blockIdx.x * 16;       // batch rows handled by this block

    for (int idx = tid; idx < RNN_DH * RNN_DH; idx += 256) {
        int k = idx >> 8, n = idx & 255;
        WhT[n * RNN_DH + k] = f2bf(Wh[(size_t)k * RNN_DH + n]);
    }
    for (int idx = tid; idx < 16 * RNN_DH; idx += 256) {
        int r = idx >> 8, k = idx & 255;
        hS[0][idx] = f2bf(hinit[(size_t)(rowBase + r) * RNN_DH + k]);
    }

    const int wave = tid >> 5, lane = tid & 31;
    const int m    = lane & 15;
    const int ks   = lane >> 4;
    const int Mb   = ks * 8;                   // C-tile row base for this lane half
    const int col0 = wave * 32 + m;            // first 16-col tile
    const int col1 = col0 + 16;                // second 16-col tile
    const float b0 = bh[col0];
    const float b1 = bh[col1];

    __syncthreads();

    int p = 0;
    for (int t = 0; t < RNN_T; ++t) {
        const unsigned short* hp = hS[p];
        unsigned short*       hn = hS[p ^ 1];

        // Issue xp loads early: no consumer until after the WMMA chains,
        // so global latency overlaps matrix math.
        float xp0[8], xp1[8];
        #pragma unroll
        for (int r = 0; r < 8; ++r) {
            size_t base = ((size_t)(rowBase + Mb + r) * RNN_T + t) * RNN_DH;
            xp0[r] = out[base + col0];
            xp1[r] = out[base + col1];
            if (t + 1 < RNN_T)
                __builtin_prefetch(&out[base + RNN_DH + col0], 0, 0);
        }

        // 4 independent WMMA chains: (c0,c1) x (low-K, high-K)
        v8f c0a, c0b, c1a, c1b;
        #pragma unroll
        for (int i = 0; i < 8; ++i) { c0a[i] = b0; c1a[i] = b1; c0b[i] = 0.0f; c1b[i] = 0.0f; }

        #pragma unroll
        for (int kb = 0; kb < 128; kb += 32) {
            // ---- low-K half ----
            {
                ushort8 alo = *reinterpret_cast<const ushort8*>(&hp[m * RNN_DH + kb + ks * 16]);
                ushort8 ahi = *reinterpret_cast<const ushort8*>(&hp[m * RNN_DH + kb + ks * 16 + 8]);
                v16bf a = pack_frag(alo, ahi);
                ushort8 p0 = *reinterpret_cast<const ushort8*>(&WhT[col0 * RNN_DH + kb + ks * 16]);
                ushort8 p1 = *reinterpret_cast<const ushort8*>(&WhT[col0 * RNN_DH + kb + ks * 16 + 8]);
                c0a = wmma_bf16(a, pack_frag(p0, p1), c0a);
                ushort8 q0 = *reinterpret_cast<const ushort8*>(&WhT[col1 * RNN_DH + kb + ks * 16]);
                ushort8 q1 = *reinterpret_cast<const ushort8*>(&WhT[col1 * RNN_DH + kb + ks * 16 + 8]);
                c1a = wmma_bf16(a, pack_frag(q0, q1), c1a);
            }
            // ---- high-K half ----
            {
                int kb2 = kb + 128;
                ushort8 alo = *reinterpret_cast<const ushort8*>(&hp[m * RNN_DH + kb2 + ks * 16]);
                ushort8 ahi = *reinterpret_cast<const ushort8*>(&hp[m * RNN_DH + kb2 + ks * 16 + 8]);
                v16bf a = pack_frag(alo, ahi);
                ushort8 p0 = *reinterpret_cast<const ushort8*>(&WhT[col0 * RNN_DH + kb2 + ks * 16]);
                ushort8 p1 = *reinterpret_cast<const ushort8*>(&WhT[col0 * RNN_DH + kb2 + ks * 16 + 8]);
                c0b = wmma_bf16(a, pack_frag(p0, p1), c0b);
                ushort8 q0 = *reinterpret_cast<const ushort8*>(&WhT[col1 * RNN_DH + kb2 + ks * 16]);
                ushort8 q1 = *reinterpret_cast<const ushort8*>(&WhT[col1 * RNN_DH + kb2 + ks * 16 + 8]);
                c1b = wmma_bf16(a, pack_frag(q0, q1), c1b);
            }
        }

        // Activation + write-out (global h overwrites xp; bf16 h -> next buffer)
        #pragma unroll
        for (int r = 0; r < 8; ++r) {
            float h0 = fast_tanh(c0a[r] + c0b[r] + xp0[r]);
            float h1 = fast_tanh(c1a[r] + c1b[r] + xp1[r]);
            size_t base = ((size_t)(rowBase + Mb + r) * RNN_T + t) * RNN_DH;
            out[base + col0] = h0;
            out[base + col1] = h1;
            int lr = Mb + r;
            hn[lr * RNN_DH + col0] = f2bf(h0);
            hn[lr * RNN_DH + col1] = f2bf(h1);
        }

        __syncthreads();   // new buffer visible to all waves; old buffer free
        p ^= 1;
    }
}

extern "C" void kernel_launch(void* const* d_in, const int* in_sizes, int n_in,
                              void* d_out, int out_size, void* d_ws, size_t ws_size,
                              hipStream_t stream) {
    const float* seq_x  = (const float*)d_in[0];   // (64, 2048, 256)
    const float* Wx     = (const float*)d_in[1];   // (256, 256)
    const float* bx     = (const float*)d_in[2];   // (1, 256)
    const float* Wh     = (const float*)d_in[3];   // (256, 256)
    const float* bh     = (const float*)d_in[4];   // (1, 256)
    const float* h_init = (const float*)d_in[5];   // (64, 256)
    float*       out    = (float*)d_out;           // (64, 2048, 256)

    // Phase 1: x_proj -> out (in-place staging). 131072 rows / 32 per block,
    // 4 column blocks of 64.
    const int rowBlocks = (RNN_N * RNN_T) / 32;    // 4096
    xproj_kernel<<<dim3(rowBlocks * 4), dim3(256), 0, stream>>>(seq_x, Wx, bx, out);

    // Phase 2: sequential scan; 4 blocks of 16 batch rows each.
    rnn_kernel<<<dim3(RNN_N / 16), dim3(256), 0, stream>>>(Wh, bh, h_init, out);
}
